// GravitationalAttention_12618613915809
// MI455X (gfx1250) — compile-verified, matrix operand order checked by reading, not run on hardware
//
#include <hip/hip_runtime.h>

#define B_   4
#define L_   2048
#define D_   512
#define H_   8
#define DH_  64
#define EPSF 1e-8f
#define KCHUNK 64                 // keys staged per async chunk (8 KB)
#define NCHUNK (L_ / KCHUNK)      // 32 chunks

typedef __attribute__((ext_vector_type(16))) _Float16 v16h;
typedef __attribute__((ext_vector_type(8)))  _Float16 v8h;
typedef __attribute__((ext_vector_type(8)))  float    v8f;
typedef __attribute__((ext_vector_type(4)))  float    v4f;

static __device__ inline v8f wmma_f16(v16h a, v16h b, v8f c) {
  // D = A(16x32 f16) * B(32x16 f16) + C(16x16 f32)
  return __builtin_amdgcn_wmma_f32_16x16x32_f16(false, a, false, b, (short)0, c,
                                                false, false);
}

// A-fragment (16x32, M x K) from row-major [rows, ld] f16; lane m = lane&15,
// per-lane K runs: [kb, kb+8) and [16+kb, 16+kb+8), kb = 8*(lane>>4).
// The B-fragment (32x16, K x N) has the symmetric layout (lane = N, VGPRs = K),
// so the SAME loader works when `base` is an N-major (K-contiguous) matrix.
static __device__ inline v16h load_frag_f16(const _Float16* base, int row0,
                                            int ld, int k0, int lane) {
  int r  = lane & 15;
  int kb = (lane >> 4) << 3;
  const _Float16* p = base + (size_t)(row0 + r) * ld + (size_t)(k0 + kb);
  v8h lo = *(const v8h*)(p);
  v8h hi = *(const v8h*)(p + 16);
  v16h f;
#pragma unroll
  for (int i = 0; i < 8; ++i) { f[i] = lo[i]; f[i + 8] = hi[i]; }
  return f;
}

// Same fragment pattern but from an f32 matrix, converting to f16 in registers.
static __device__ inline v16h load_frag_f32cvt(const float* base, int row0,
                                               int ld, int k0, int lane) {
  int r  = lane & 15;
  int kb = (lane >> 4) << 3;
  const float* p = base + (size_t)(row0 + r) * ld + (size_t)(k0 + kb);
  v4f a0 = *(const v4f*)(p);
  v4f a1 = *(const v4f*)(p + 4);
  v4f b0 = *(const v4f*)(p + 16);
  v4f b1 = *(const v4f*)(p + 20);
  v16h f;
#pragma unroll
  for (int i = 0; i < 4; ++i) {
    f[i]      = (_Float16)a0[i];
    f[i + 4]  = (_Float16)a1[i];
    f[i + 8]  = (_Float16)b0[i];
    f[i + 12] = (_Float16)b1[i];
  }
  return f;
}

static __device__ inline float softplus_f(float x) {
  return (x > 20.0f) ? x : log1pf(expf(x));
}

// ---------------------------------------------------------------------------
// 1) P = X @ W^T + bias, written as f16. mode 0/1: [B,H,L,DH]; mode 2: [B,H,DH,L]
// One wave computes a 16x16 output tile. Grid: 512 (M) x 32 (N) tiles.
// ---------------------------------------------------------------------------
__global__ void __launch_bounds__(128) proj_kernel(const float* __restrict__ X,
                                                   const float* __restrict__ W,
                                                   const float* __restrict__ bias,
                                                   _Float16* __restrict__ out,
                                                   int mode) {
  int lane = threadIdx.x & 31;
  int w    = blockIdx.x * 4 + (threadIdx.x >> 5);   // 0..16383
  int mt   = w >> 5;                                // 0..511
  int nt   = w & 31;                                // 0..31
  int m0 = mt * 16, n0 = nt * 16;

  v8f c = {};
#pragma unroll 4
  for (int k = 0; k < D_; k += 32) {
    v16h a = load_frag_f32cvt(X, m0, D_, k, lane);  // A: rows of X
    v16h b = load_frag_f32cvt(W, n0, D_, k, lane);  // B: W is [n][k], K-contiguous
    c = wmma_f16(a, b, c);
  }

  int nl   = lane & 15;
  int mrow = (lane >> 4) * 8;
  int ncol = n0 + nl;
  float bv = bias[ncol];
  int hh = ncol >> 6;        // head
  int dh = ncol & 63;        // dim within head
#pragma unroll
  for (int r = 0; r < 8; ++r) {
    int row = m0 + mrow + r;          // 0..8191
    int bb  = row >> 11;              // batch
    int ll  = row & 2047;             // position
    float v = c[r] + bv;
    if (mode == 2)
      out[((size_t)(bb * H_ + hh) * DH_ + dh) * L_ + ll] = (_Float16)v;   // V^T
    else
      out[((size_t)(bb * H_ + hh) * L_ + ll) * DH_ + dh] = (_Float16)v;   // Q,K
  }
}

// ---------------------------------------------------------------------------
// 2) masses (softplus(X @ Wm^T + bm)) and squared norms of projected Q/K.
// One thread per (b,l,h): 65536 threads.
// ---------------------------------------------------------------------------
__global__ void mass_kernel(const float* __restrict__ qin,
                            const float* __restrict__ kin,
                            const float* __restrict__ Wmq,
                            const float* __restrict__ bmq,
                            const float* __restrict__ Wmk,
                            const float* __restrict__ bmk,
                            const _Float16* __restrict__ qf,
                            const _Float16* __restrict__ kf,
                            float* __restrict__ qm, float* __restrict__ km,
                            float* __restrict__ q2, float* __restrict__ k2) {
  int tid = blockIdx.x * blockDim.x + threadIdx.x;   // 0..65535
  int h = tid & 7;
  int l = (tid >> 3) & 2047;
  int b = tid >> 14;

  const float* qr = qin + (size_t)(b * L_ + l) * D_;
  const float* kr = kin + (size_t)(b * L_ + l) * D_;
  const float* wq = Wmq + (size_t)h * D_;
  const float* wk = Wmk + (size_t)h * D_;
  float sq = bmq[h], sk = bmk[h];
  for (int i = 0; i < D_; i += 4) {
    v4f a = *(const v4f*)(qr + i);
    v4f c = *(const v4f*)(kr + i);
    v4f u = *(const v4f*)(wq + i);
    v4f v = *(const v4f*)(wk + i);
    sq += a[0]*u[0] + a[1]*u[1] + a[2]*u[2] + a[3]*u[3];
    sk += c[0]*v[0] + c[1]*v[1] + c[2]*v[2] + c[3]*v[3];
  }
  size_t idx = (size_t)(b * H_ + h) * L_ + l;
  qm[idx] = softplus_f(sq);
  km[idx] = softplus_f(sk);

  const _Float16* qp = qf + idx * DH_;
  const _Float16* kp = kf + idx * DH_;
  float aq = 0.f, ak = 0.f;
  for (int d = 0; d < DH_; d += 8) {
    v8h x = *(const v8h*)(qp + d);
    v8h y = *(const v8h*)(kp + d);
#pragma unroll
    for (int j = 0; j < 8; ++j) {
      float fx = (float)x[j], fy = (float)y[j];
      aq += fx * fx;
      ak += fy * fy;
    }
  }
  q2[idx] = aq;
  k2[idx] = ak;
}

// ---------------------------------------------------------------------------
// 3) Gravity attention. One wave owns 16 q-rows of one (b,h); the 4 waves of a
// block share the SAME (b,h), so the block cooperatively stages the K stream
// through LDS with double-buffered global_load_async_to_lds_b128 (ASYNCcnt).
// Emits unnormalized attention (f32, non-temporal) to d_out, row sums to ws,
// and the already-normalized x tile (f16) via a second WMMA chain (gravity@V).
// ---------------------------------------------------------------------------
__global__ void __launch_bounds__(128) gravity_kernel(
    const _Float16* __restrict__ qf, const _Float16* __restrict__ kf,
    const _Float16* __restrict__ vt, const float* __restrict__ qm,
    const float* __restrict__ km, const float* __restrict__ q2,
    const float* __restrict__ k2, const float* __restrict__ G,
    float* __restrict__ attn, float* __restrict__ rowsum,
    _Float16* __restrict__ xf) {
  __shared__ __align__(16) _Float16 kbuf[2][KCHUNK * DH_];  // 2 x 8 KB K chunks
  __shared__ __align__(16) _Float16 ldsG[4][16][32];        // per-wave restripe
  int tid  = threadIdx.x;
  int lane = tid & 31;
  int wv   = tid >> 5;
  int w    = blockIdx.x * 4 + wv;      // 0..4095
  int qt   = w & 127;
  int bh   = w >> 7;                   // 0..31 (uniform across the block)
  int h    = bh & 7;
  int bb   = bh >> 3;
  int qbase = qt * 16;

  const _Float16* qbh = qf + (size_t)bh * L_ * DH_;
  const _Float16* kbh = kf + (size_t)bh * L_ * DH_;
  const _Float16* vbh = vt + (size_t)bh * DH_ * L_;
  float Gh = G[h];

  // block-cooperative async copy of one 64-key chunk into kbuf[nb]
  auto issue_chunk = [&](int c, int nb) {
    unsigned long long src = (unsigned long long)(size_t)(kbh + (size_t)c * KCHUNK * DH_);
    unsigned ldsbase = (unsigned)(unsigned long long)(size_t)(&kbuf[nb][0]);
#pragma unroll
    for (int j = 0; j < 4; ++j) {
      unsigned off   = (unsigned)((tid + j * 128) * 16);
      unsigned daddr = ldsbase + off;
      unsigned long long gaddr = src + off;
      asm volatile("global_load_async_to_lds_b128 %0, %1, off"
                   :: "v"(daddr), "v"(gaddr) : "memory");
    }
  };

  // Q A-fragments (K = DH = 64 -> two fragments), reused for the whole sweep.
  v16h aq0 = load_frag_f16(qbh, qbase, DH_, 0, lane);
  v16h aq1 = load_frag_f16(qbh, qbase, DH_, 32, lane);

  int nl   = lane & 15;
  int mrow = (lane >> 4) * 8;
  size_t qko = (size_t)bh * L_ + qbase;
  float q2r[8], qmr[8];
#pragma unroll
  for (int r = 0; r < 8; ++r) {
    q2r[r] = q2[qko + mrow + r];
    qmr[r] = Gh * qm[qko + mrow + r];    // fold G into the q mass
  }

  v8f accv[4] = {};                      // gravity @ V accumulators (DH = 4x16)
  float rs[8] = {};
  float* attn_q0 = attn + ((size_t)bh * L_ + qbase) * L_;

  issue_chunk(0, 0);
  for (int c = 0; c < NCHUNK; ++c) {
    int cur = c & 1;
    if (c + 1 < NCHUNK) {
      issue_chunk(c + 1, cur ^ 1);
      // each thread just issued 4 more; in-order completion => chunk c resident
      asm volatile("s_wait_asynccnt 4" ::: "memory");
    } else {
      asm volatile("s_wait_asynccnt 0" ::: "memory");
    }
    __syncthreads();                       // whole block sees chunk c in LDS
    const _Float16* kchunk = &kbuf[cur][0];

#pragma unroll
    for (int kk = 0; kk < KCHUNK; kk += 32) {
      int kt = c * KCHUNK + kk;
#pragma unroll
      for (int t = 0; t < 2; ++t) {
        int klocal = kk + t * 16;
        v16h bk0 = load_frag_f16(kchunk, klocal, DH_, 0, lane);   // from LDS
        v16h bk1 = load_frag_f16(kchunk, klocal, DH_, 32, lane);
        v8f s = {};
        s = wmma_f16(aq0, bk0, s);
        s = wmma_f16(aq1, bk1, s);

        int key = kt + t * 16 + nl;
        float k2v = k2[(size_t)bh * L_ + key];
        float kmv = km[(size_t)bh * L_ + key];
#pragma unroll
        for (int r = 0; r < 8; ++r) {
          float dist = fmaxf(q2r[r] + k2v - 2.0f * s[r], 0.0f) + EPSF;
          float g = qmr[r] * kmv / dist;
          // non-temporal: do not let the 537MB attn stream evict L2 working set
          __builtin_nontemporal_store(g, &attn_q0[(size_t)(mrow + r) * L_ + key]);
          rs[r] += g;
          ldsG[wv][mrow + r][t * 16 + nl] = (_Float16)g;  // C/D -> A restripe
        }
      }
      asm volatile("s_wait_dscnt 0" ::: "memory");
      v16h ag = load_frag_f16(&ldsG[wv][0][0], 0, 32, 0, lane);
#pragma unroll
      for (int nt = 0; nt < 4; ++nt) {
        v16h bv = load_frag_f16(vbh, nt * 16, L_, kt, lane); // V^T: K-contiguous
        accv[nt] = wmma_f16(ag, bv, accv[nt]);
      }
    }
    __syncthreads();   // all waves done reading kbuf[cur] before it is refilled
  }

  // reduce row sums across the 16 N-lanes of the C/D layout
#pragma unroll
  for (int mask = 1; mask < 16; mask <<= 1)
#pragma unroll
    for (int r = 0; r < 8; ++r) rs[r] += __shfl_xor(rs[r], mask, 32);

  if (nl == 0) {
#pragma unroll
    for (int r = 0; r < 8; ++r) rowsum[qko + mrow + r] = rs[r];
  }

  float inv[8];
#pragma unroll
  for (int r = 0; r < 8; ++r) inv[r] = 1.0f / (rs[r] + EPSF);

#pragma unroll
  for (int nt = 0; nt < 4; ++nt) {
#pragma unroll
    for (int r = 0; r < 8; ++r) {
      int l = qbase + mrow + r;
      int col = h * DH_ + nt * 16 + nl;
      xf[(size_t)(bb * L_ + l) * D_ + col] = (_Float16)(accv[nt][r] * inv[r]);
    }
  }
}

// ---------------------------------------------------------------------------
// 4) Streaming normalization of the attention output: attn /= (rowsum + eps)
// Pure bandwidth pass: non-temporal both ways.
// ---------------------------------------------------------------------------
__global__ void norm_kernel(float* __restrict__ attn,
                            const float* __restrict__ rowsum) {
  size_t i = (size_t)blockIdx.x * blockDim.x + threadIdx.x;  // float4 index
  v4f* a4 = (v4f*)attn;
  size_t row = i >> 9;                       // (i*4)/2048
  float inv = 1.0f / (rowsum[row] + EPSF);
  v4f g = __builtin_nontemporal_load(&a4[i]);
  __builtin_nontemporal_store(g * inv, &a4[i]);
}

// ---------------------------------------------------------------------------
// 5) output projection: out = x @ Wo^T + bo  (f16 WMMA, f32 result)
// ---------------------------------------------------------------------------
__global__ void __launch_bounds__(128) outproj_kernel(
    const _Float16* __restrict__ Xh, const float* __restrict__ Wo,
    const float* __restrict__ bo, float* __restrict__ out) {
  int lane = threadIdx.x & 31;
  int w    = blockIdx.x * 4 + (threadIdx.x >> 5);
  int mt   = w >> 5;
  int nt   = w & 31;
  int m0 = mt * 16, n0 = nt * 16;

  v8f c = {};
#pragma unroll 4
  for (int k = 0; k < D_; k += 32) {
    v16h a = load_frag_f16(Xh, m0, D_, k, lane);
    v16h b = load_frag_f32cvt(Wo, n0, D_, k, lane);
    c = wmma_f16(a, b, c);
  }
  int nl   = lane & 15;
  int mrow = (lane >> 4) * 8;
  int ncol = n0 + nl;
  float bv = bo[ncol];
#pragma unroll
  for (int r = 0; r < 8; ++r) {
    int row = m0 + mrow + r;
    out[(size_t)row * D_ + ncol] = c[r] + bv;
  }
}

// ---------------------------------------------------------------------------
extern "C" void kernel_launch(void* const* d_in, const int* in_sizes, int n_in,
                              void* d_out, int out_size, void* d_ws,
                              size_t ws_size, hipStream_t stream) {
  const float* query = (const float*)d_in[0];
  const float* key_  = (const float*)d_in[1];
  const float* value = (const float*)d_in[2];
  const float* Wq    = (const float*)d_in[3];
  const float* bq    = (const float*)d_in[4];
  const float* Wk    = (const float*)d_in[5];
  const float* bk    = (const float*)d_in[6];
  const float* Wv    = (const float*)d_in[7];
  const float* bv    = (const float*)d_in[8];
  const float* Wo    = (const float*)d_in[9];
  const float* bo    = (const float*)d_in[10];
  const float* Wmq   = (const float*)d_in[11];
  const float* bmq   = (const float*)d_in[12];
  const float* Wmk   = (const float*)d_in[13];
  const float* bmk   = (const float*)d_in[14];
  const float* G     = (const float*)d_in[15];

  char* ws = (char*)d_ws;
  const size_t HALF_MAT = (size_t)B_ * L_ * D_ * sizeof(_Float16);   // 8 MB
  _Float16* qf = (_Float16*)(ws);
  _Float16* kf = (_Float16*)(ws + HALF_MAT);
  _Float16* vt = (_Float16*)(ws + 2 * HALF_MAT);
  _Float16* xf = (_Float16*)(ws + 3 * HALF_MAT);
  float* qm     = (float*)(ws + 4 * HALF_MAT);
  float* km     = qm + (size_t)B_ * H_ * L_;
  float* q2     = km + (size_t)B_ * H_ * L_;
  float* k2     = q2 + (size_t)B_ * H_ * L_;
  float* rowsum = k2 + (size_t)B_ * H_ * L_;

  float* out  = (float*)d_out;
  float* attn = out + (size_t)B_ * L_ * D_;   // attention follows output

  dim3 blk(128);
  // 1) projections (Q, K row-major per head; V transposed per head)
  proj_kernel<<<4096, blk, 0, stream>>>(query, Wq, bq, qf, 0);
  proj_kernel<<<4096, blk, 0, stream>>>(key_, Wk, bk, kf, 1);
  proj_kernel<<<4096, blk, 0, stream>>>(value, Wv, bv, vt, 2);
  // 2) masses + squared norms
  mass_kernel<<<256, 256, 0, stream>>>(query, key_, Wmq, bmq, Wmk, bmk, qf, kf,
                                       qm, km, q2, k2);
  // 3) gravity sweep: unnormalized attn + rowsums + normalized x (f16)
  gravity_kernel<<<1024, blk, 0, stream>>>(qf, kf, vt, qm, km, q2, k2, G, attn,
                                           rowsum, xf);
  // 4) normalize attention in place
  norm_kernel<<<131072, 256, 0, stream>>>(attn, rowsum);
  // 5) output projection
  outproj_kernel<<<4096, blk, 0, stream>>>(xf, Wo, bo, out);
}
